// MetaMorph_76544907149717
// MI455X (gfx1250) — matrix operand order repaired and checked by reading, not compiled, not run on hardware
//
#include <hip/hip_runtime.h>

#define DEV __device__ __forceinline__

typedef __attribute__((ext_vector_type(16))) _Float16 v16h;
typedef __attribute__((ext_vector_type(8)))  _Float16 v8h;
typedef __attribute__((ext_vector_type(8)))  float    v8f;

constexpr int   NV      = 262144;          // 64^3 voxels
constexpr int   LSTEPS  = 10;
constexpr float SLOPE   = 0.01f;
constexpr float INV_L   = 0.1f;
constexpr float MU2_L   = 0.02f * 0.02f * 0.1f;   // MU^2 / L

// Normalized Gaussian taps, sigma=2, KS=13 (exp(-r^2/8)/sum)
__constant__ float GK[13] = {
  0.0022182f, 0.0087731f, 0.0270231f, 0.0648251f, 0.1211095f, 0.1762131f,
  0.1996757f,
  0.1762131f, 0.1211095f, 0.0648251f, 0.0270231f, 0.0087731f, 0.0022182f };

// ---------------------------------------------------------------------------
// WMMA implicit-GEMM 3x3x3 conv.  Activations: f16 channel-last [NV][CINPAD].
// Weights pre-swizzled to A-fragment order:
//   [tap][mtile][kblock][lane(32)][16 halfs]  -> one v16h load per fragment.
// One wave = (MTILES*16) Cout x (NTILES*16) voxels; NTILES=4 -> a full x-row,
// so each A (weight) fragment is reused across 4 B fragments (1 load / WMMA).
// ---------------------------------------------------------------------------
template<int MTILES, int NTILES, int KBLOCKS, int CINPAD, bool RELU, bool F16OUT>
__global__ __launch_bounds__(128) void conv_wmma_k(
    const _Float16* __restrict__ act,
    const _Float16* __restrict__ wgt,
    _Float16* __restrict__ outh,   // [NV][64] when F16OUT
    float*    __restrict__ outf)   // [NV]     when !F16OUT (Cout=0 only)
{
  const int lane  = threadIdx.x & 31;
  const int wave  = threadIdx.x >> 5;
  const int group = blockIdx.x * 4 + wave;
  const int base  = group * (16 * NTILES);      // first voxel of the wave's row
  const int z  = base >> 12, y = (base >> 6) & 63, x0 = base & 63;
  const int n  = lane & 15;                     // N column within a 16-tile
  const int hi = lane >> 4;                     // lane-group 0/1
  constexpr int TAPW = MTILES * KBLOCKS * 512;  // halfs per tap of weights

  v8f acc[MTILES][NTILES] = {};

  for (int tap = 0; tap < 27; ++tap) {
    const int dz = tap / 9 - 1, dy = (tap / 3) % 3 - 1, dx = tap % 3 - 1;
    const int zz = z + dz, yy = y + dy;
    const bool rowok = ((unsigned)zz < 64u) && ((unsigned)yy < 64u);
    const long rowoff = (long)(((zz & 63) * 64 + (yy & 63)) * 64) * CINPAD;
    const _Float16* wtap = wgt + (long)tap * TAPW;
    if (tap < 26)   // prefetch next tap's weights (global_prefetch_b8)
      __builtin_prefetch(wtap + TAPW + lane * (TAPW / 32), 0, 3);

#pragma unroll
    for (int kb = 0; kb < KBLOCKS; ++kb) {
      // A fragments: one contiguous 32B per lane, already in WMMA layout
      v16h a[MTILES];
#pragma unroll
      for (int mt = 0; mt < MTILES; ++mt)
        a[mt] = *(const v16h*)(wtap + (((mt * KBLOCKS + kb) * 32 + lane) << 4));

#pragma unroll
      for (int nt = 0; nt < NTILES; ++nt) {
        // B fragment: lane holds 16 contiguous halfs (K = kb*32 + hi*16 + j)
        const int xx = x0 + nt * 16 + n + dx;
        const bool ok = rowok && ((unsigned)xx < 64u);
        v16h b = {};
        if (ok) b = *(const v16h*)(act + rowoff + (long)(xx & 63) * CINPAD
                                       + kb * 32 + hi * 16);
#pragma unroll
        for (int mt = 0; mt < MTILES; ++mt)
          acc[mt][nt] = __builtin_amdgcn_wmma_f32_16x16x32_f16(
              false, a[mt], false, b, (short)0, acc[mt][nt], false, false);
      }
    }
  }

  if constexpr (F16OUT) {
#pragma unroll
    for (int nt = 0; nt < NTILES; ++nt) {
      const long vb = (long)(base + nt * 16 + n) * 64;
#pragma unroll
      for (int mt = 0; mt < MTILES; ++mt) {
        v8h o;
#pragma unroll
        for (int r = 0; r < 8; ++r) {         // Cout = mt*16 + hi*8 + r
          float v = acc[mt][nt][r];
          if (RELU) v = v > 0.f ? v : v * SLOPE;
          o[r] = (_Float16)v;
        }
        *(v8h*)(outh + vb + mt * 16 + hi * 8) = o;
      }
    }
  } else {
#pragma unroll
    for (int nt = 0; nt < NTILES; ++nt) {
      float v = acc[0][nt][0];                // M=0 lives in elem 0, lanes 0..15
      if (RELU) v = v > 0.f ? v : v * SLOPE;
      if (lane < 16) outf[base + nt * 16 + n] = v;
    }
  }
}

// ---------------------------------------------------------------------------
// Weight repack f32 [step][coutR][cinR][3][3][3]
//   -> f16 fragment order [step][27][mtiles][kblocks][lane(32)][16]
// lane l, half j: A[M = mt*16 + l%16][K = kb*32 + j + 16*(l/16)]
// ---------------------------------------------------------------------------
__global__ void wpack_k(const float* __restrict__ w, _Float16* __restrict__ dst,
                        int coutR, int cinR, int mtiles, int kblocks, long total)
{
  long t = (long)blockIdx.x * blockDim.x + threadIdx.x;
  if (t >= total) return;
  int j    = (int)(t & 15);
  int lane = (int)((t >> 4) & 31);
  long r   = t >> 9;
  int kb   = (int)(r % kblocks); r /= kblocks;
  int mt   = (int)(r % mtiles);  r /= mtiles;
  int tap  = (int)(r % 27);      int step = (int)(r / 27);
  int cout = mt * 16 + (lane & 15);
  int cin  = kb * 32 + j + (lane >> 4) * 16;
  float val = 0.f;
  if (cout < coutR && cin < cinR)
    val = w[(((long)step * coutR + cout) * cinR + cin) * 27 + tap];
  dst[t] = (_Float16)val;
}

// ---------------------------------------------------------------------------
// Pointwise / sampling kernels (f32)
// ---------------------------------------------------------------------------
__global__ void init_k(const float* __restrict__ source, const float* __restrict__ z0,
                       float* residual, float* image, float* resdef, float* phi,
                       float* resid0_out)
{
  int vox = blockIdx.x * blockDim.x + threadIdx.x;
  int z = vox >> 12, y = (vox >> 6) & 63, x = vox & 63;
  float r = z0[vox];
  residual[vox]   = r;
  resid0_out[vox] = r;
  image[vox]      = source[vox];
  resdef[vox]     = 0.f;
  phi[vox]          = -1.f + x * (2.f / 63.f);   // grid x -> W
  phi[NV + vox]     = -1.f + y * (2.f / 63.f);   // grid y -> H
  phi[2 * NV + vox] = -1.f + z * (2.f / 63.f);   // grid z -> D
}

__global__ void grad_v_k(const float* __restrict__ img, const float* __restrict__ res,
                         float* __restrict__ v0, float* __restrict__ gout)
{
  int vox = blockIdx.x * blockDim.x + threadIdx.x;
  int z = vox >> 12, y = (vox >> 6) & 63, x = vox & 63;
  int xm = x > 0 ? x - 1 : 0, xp = x < 63 ? x + 1 : 63;
  int ym = y > 0 ? y - 1 : 0, yp = y < 63 ? y + 1 : 63;
  int zm = z > 0 ? z - 1 : 0, zp = z < 63 ? z + 1 : 63;
  float gx = 0.5f * (img[(z * 64 + y) * 64 + xp] - img[(z * 64 + y) * 64 + xm]);
  float gy = 0.5f * (img[(z * 64 + yp) * 64 + x] - img[(z * 64 + ym) * 64 + x]);
  float gz = 0.5f * (img[(zp * 64 + y) * 64 + x] - img[(zm * 64 + y) * 64 + x]);
  gout[vox] = gx; gout[NV + vox] = gy; gout[2 * NV + vox] = gz;
  float r = res[vox];
  v0[vox] = -r * gx; v0[NV + vox] = -r * gy; v0[2 * NV + vox] = -r * gz;
}

template<int AXIS>  // 0:D 1:H 2:W, zero ('constant') border
__global__ void smooth_k(float* __restrict__ dst, const float* __restrict__ src)
{
  int t = blockIdx.x * blockDim.x + threadIdx.x;
  int c = t >> 18;                     // t / NV
  int vox = t & (NV - 1);
  int z = vox >> 12, y = (vox >> 6) & 63, x = vox & 63;
  int coord  = AXIS == 0 ? z : (AXIS == 1 ? y : x);
  int stride = AXIS == 0 ? 4096 : (AXIS == 1 ? 64 : 1);
  const float* p = src + c * NV + vox;
  float s = 0.f;
#pragma unroll
  for (int k = 0; k < 13; ++k) {
    int cc = coord + k - 6;
    if (cc >= 0 && cc < 64) s += GK[k] * p[(k - 6) * stride];
  }
  dst[t] = s;
}

__global__ void fields_out_k(const float* __restrict__ vf, float* __restrict__ dst)
{
  int t = blockIdx.x * blockDim.x + threadIdx.x;   // [W][H][D][3]
  int c = t % 3, s = t / 3;
  int w = s >> 12, h = (s >> 6) & 63, d = s & 63;
  dst[t] = vf[c * NV + (d * 64 + h) * 64 + w];
}

__global__ void pack_in_k(const float* __restrict__ residual, const float* __restrict__ image,
                          const float* __restrict__ target, _Float16* __restrict__ dst)
{
  int vox = blockIdx.x * blockDim.x + threadIdx.x;
  _Float16* p = dst + (long)vox * 32;
  v8h zc = {};
  v8h a  = {};
  a[0] = (_Float16)residual[vox];
  a[1] = (_Float16)image[vox];
  a[2] = (_Float16)target[vox];
  ((v8h*)p)[0] = a; ((v8h*)p)[1] = zc; ((v8h*)p)[2] = zc; ((v8h*)p)[3] = zc;
}

__global__ void resid_update_k(float* __restrict__ residual, const float* __restrict__ fout,
                               float* __restrict__ res_out)
{
  int vox = blockIdx.x * blockDim.x + threadIdx.x;
  float r = residual[vox] + fout[vox] * INV_L;
  residual[vox] = r;
  res_out[vox]  = r;
}

DEV float trilerp(const float* __restrict__ img, float X, float Y, float Z,
                  int sx, int sy, int sz)
{
  float xf = floorf(X), yf = floorf(Y), zf = floorf(Z);
  int x0 = (int)xf, y0 = (int)yf, z0 = (int)zf;
  int x1 = x0 + 1; if (x1 > 63) x1 = 63;
  int y1 = y0 + 1; if (y1 > 63) y1 = 63;
  int z1 = z0 + 1; if (z1 > 63) z1 = 63;
  float wx = X - xf, wy = Y - yf, wz = Z - zf;
  float c000 = img[z0 * sz + y0 * sy + x0 * sx];
  float c001 = img[z0 * sz + y0 * sy + x1 * sx];
  float c010 = img[z0 * sz + y1 * sy + x0 * sx];
  float c011 = img[z0 * sz + y1 * sy + x1 * sx];
  float c100 = img[z1 * sz + y0 * sy + x0 * sx];
  float c101 = img[z1 * sz + y0 * sy + x1 * sx];
  float c110 = img[z1 * sz + y1 * sy + x0 * sx];
  float c111 = img[z1 * sz + y1 * sy + x1 * sx];
  float c00 = c000 + (c001 - c000) * wx;
  float c01 = c010 + (c011 - c010) * wx;
  float c10 = c100 + (c101 - c100) * wx;
  float c11 = c110 + (c111 - c110) * wx;
  float c0 = c00 + (c01 - c00) * wy;
  float c1 = c10 + (c11 - c10) * wy;
  return c0 + (c1 - c0) * wz;
}

// deformation = id_grid - field/L at output voxel (d,h,w); field[d,h,w,c]=v[c][w,h,d]
DEV void defpos(const float* __restrict__ vf, int d, int h, int w,
                float& X, float& Y, float& Z)
{
  const int vox2 = (w * 64 + h) * 64 + d;
  float gx = (-1.f + w * (2.f / 63.f)) - vf[vox2] * INV_L;
  float gy = (-1.f + h * (2.f / 63.f)) - vf[NV + vox2] * INV_L;
  float gz = (-1.f + d * (2.f / 63.f)) - vf[2 * NV + vox2] * INV_L;
  X = fminf(fmaxf((gx + 1.f) * 31.5f, 0.f), 63.f);
  Y = fminf(fmaxf((gy + 1.f) * 31.5f, 0.f), 63.f);
  Z = fminf(fmaxf((gz + 1.f) * 31.5f, 0.f), 63.f);
}

__global__ void warp_resdef_k(const float* __restrict__ rin, float* __restrict__ rout,
                              const float* __restrict__ vf, const float* __restrict__ residual,
                              int doWarp)
{
  int vox = blockIdx.x * blockDim.x + threadIdx.x;
  int d = vox >> 12, h = (vox >> 6) & 63, w = vox & 63;
  float val;
  if (doWarp) {
    float X, Y, Z; defpos(vf, d, h, w, X, Y, Z);
    val = trilerp(rin, X, Y, Z, 1, 64, 4096);
  } else {
    val = rin[vox];
  }
  rout[vox] = val + residual[vox];
}

// phi = T(grid_sample(T(phi), deformation)); T = (0,4,3,2,1) transpose
__global__ void warp_phi_k(const float* __restrict__ pin, float* __restrict__ pout,
                           const float* __restrict__ vf)
{
  int vox = blockIdx.x * blockDim.x + threadIdx.x;
  int ds = vox >> 12, hs = (vox >> 6) & 63, ws = vox & 63;   // sample coords
  float X, Y, Z; defpos(vf, ds, hs, ws, X, Y, Z);
  int widx = (ws * 64 + hs) * 64 + ds;                       // transposed write
  pout[widx]          = trilerp(pin,          X, Y, Z, 4096, 64, 1);  // transposed gather
  pout[NV + widx]     = trilerp(pin + NV,     X, Y, Z, 4096, 64, 1);
  pout[2 * NV + widx] = trilerp(pin + 2 * NV, X, Y, Z, 4096, 64, 1);
}

__global__ void mask_image_k(const float* __restrict__ phi, const float* __restrict__ rd,
                             const float* __restrict__ src, const float* __restrict__ seg,
                             float* __restrict__ image)
{
  int vox = blockIdx.x * blockDim.x + threadIdx.x;
  float X = fminf(fmaxf((phi[vox]          + 1.f) * 31.5f, 0.f), 63.f);
  float Y = fminf(fmaxf((phi[NV + vox]     + 1.f) * 31.5f, 0.f), 63.f);
  float Z = fminf(fmaxf((phi[2 * NV + vox] + 1.f) * 31.5f, 0.f), 63.f);
  float m = trilerp(seg, X, Y, Z, 1, 64, 4096);
  float s = trilerp(src, X, Y, Z, 1, 64, 4096);
  image[vox] = s + rd[vox] * MU2_L * m;
}

// ---------------------------------------------------------------------------
extern "C" void kernel_launch(void* const* d_in, const int* in_sizes, int n_in,
                              void* d_out, int out_size, void* d_ws, size_t ws_size,
                              hipStream_t stream)
{
  (void)in_sizes; (void)n_in; (void)out_size;
  const float* source = (const float*)d_in[0];
  const float* target = (const float*)d_in[1];
  const float* seg    = (const float*)d_in[2];
  const float* z0     = (const float*)d_in[3];
  const float* W1     = (const float*)d_in[4];
  const float* W2     = (const float*)d_in[5];
  const float* W3     = (const float*)d_in[6];
  float* out = (float*)d_out;

  char* p = (char*)d_ws;
  auto take = [&](size_t bytes) { char* r = p; p += (bytes + 255) & ~(size_t)255; return r; };
  _Float16* inpack = (_Float16*)take((size_t)NV * 32 * 2);
  _Float16* hid1   = (_Float16*)take((size_t)NV * 64 * 2);
  _Float16* hid2   = (_Float16*)take((size_t)NV * 64 * 2);
  float* fout     = (float*)take((size_t)NV * 4);
  float* residual = (float*)take((size_t)NV * 4);
  float* image    = (float*)take((size_t)NV * 4);
  float* rdA      = (float*)take((size_t)NV * 4);
  float* rdB      = (float*)take((size_t)NV * 4);
  float* phiA     = (float*)take((size_t)3 * NV * 4);
  float* phiB     = (float*)take((size_t)3 * NV * 4);
  float* smA      = (float*)take((size_t)3 * NV * 4);
  float* smB      = (float*)take((size_t)3 * NV * 4);
  _Float16* w1p = (_Float16*)take((size_t)LSTEPS * 27 * 4 * 1 * 512 * 2);
  _Float16* w2p = (_Float16*)take((size_t)LSTEPS * 27 * 4 * 2 * 512 * 2);
  _Float16* w3p = (_Float16*)take((size_t)LSTEPS * 27 * 1 * 2 * 512 * 2);
  if ((size_t)(p - (char*)d_ws) > ws_size) return;   // workspace too small

  const long o_fields = NV;
  const long o_grads  = o_fields + (long)LSTEPS * 3 * NV;
  const long o_res    = o_grads  + (long)LSTEPS * 3 * NV;
  const long o_resdef = o_res    + (long)(LSTEPS + 1) * NV;

  const long t1 = (long)LSTEPS * 27 * 4 * 1 * 512;
  const long t2 = (long)LSTEPS * 27 * 4 * 2 * 512;
  const long t3 = (long)LSTEPS * 27 * 1 * 2 * 512;
  wpack_k<<<(int)((t1 + 255) / 256), 256, 0, stream>>>(W1, w1p, 50, 3, 4, 1, t1);
  wpack_k<<<(int)((t2 + 255) / 256), 256, 0, stream>>>(W2, w2p, 50, 50, 4, 2, t2);
  wpack_k<<<(int)((t3 + 255) / 256), 256, 0, stream>>>(W3, w3p, 1, 50, 1, 2, t3);
  init_k<<<NV / 256, 256, 0, stream>>>(source, z0, residual, image, rdA, phiA, out + o_res);

  float* rdIn = rdA;  float* rdOut = rdB;
  float* phIn = phiA; float* phOut = phiB;

  for (int i = 0; i < LSTEPS; ++i) {
    grad_v_k<<<NV / 256, 256, 0, stream>>>(image, residual, smA,
                                           out + o_grads + (long)i * 3 * NV);
    smooth_k<0><<<(3 * NV) / 256, 256, 0, stream>>>(smB, smA);
    smooth_k<1><<<(3 * NV) / 256, 256, 0, stream>>>(smA, smB);
    smooth_k<2><<<(3 * NV) / 256, 256, 0, stream>>>(smB, smA);   // vf = smB
    fields_out_k<<<(3 * NV) / 256, 256, 0, stream>>>(smB, out + o_fields + (long)i * 3 * NV);

    pack_in_k<<<NV / 256, 256, 0, stream>>>(residual, image, target, inpack);
    // 64 voxels per wave, 256 per block -> NV/256 blocks
    conv_wmma_k<4, 4, 1, 32, true,  true ><<<NV / 256, 128, 0, stream>>>(
        inpack, w1p + (long)i * 27 * 4 * 1 * 512, hid1, nullptr);
    conv_wmma_k<4, 4, 2, 64, true,  true ><<<NV / 256, 128, 0, stream>>>(
        hid1,   w2p + (long)i * 27 * 4 * 2 * 512, hid2, nullptr);
    conv_wmma_k<1, 4, 2, 64, false, false><<<NV / 256, 128, 0, stream>>>(
        hid2,   w3p + (long)i * 27 * 1 * 2 * 512, nullptr, fout);

    resid_update_k<<<NV / 256, 256, 0, stream>>>(residual, fout,
                                                 out + o_res + (long)(i + 1) * NV);
    warp_resdef_k<<<NV / 256, 256, 0, stream>>>(rdIn, rdOut, smB, residual, i > 0 ? 1 : 0);
    warp_phi_k<<<NV / 256, 256, 0, stream>>>(phIn, phOut, smB);
    mask_image_k<<<NV / 256, 256, 0, stream>>>(phOut, rdOut, source, seg, image);

    { float* t = rdIn; rdIn = rdOut; rdOut = t; }
    { float* t = phIn; phIn = phOut; phOut = t; }
  }

  hipMemcpyAsync(out,            image, (size_t)NV * 4, hipMemcpyDeviceToDevice, stream);
  hipMemcpyAsync(out + o_resdef, rdIn,  (size_t)NV * 4, hipMemcpyDeviceToDevice, stream);
}
